// ConversationAwareHGTLayer_19413252177997
// MI455X (gfx1250) — matrix operand order repaired and checked by reading, not compiled, not run on hardware
//
#include <hip/hip_runtime.h>

#define NU 50000
#define NC 100000
#define E1N 500000
#define E2N 250000

typedef __attribute__((ext_vector_type(16))) __bf16 v16bf;
typedef __attribute__((ext_vector_type(8)))  float  v8f;
typedef unsigned int u32x4 __attribute__((ext_vector_type(4)));
typedef int          i32x4 __attribute__((ext_vector_type(4)));
typedef int          i32x8 __attribute__((ext_vector_type(8)));

#if defined(__has_builtin)
#if __has_builtin(__builtin_amdgcn_tensor_load_to_lds) && __has_builtin(__builtin_amdgcn_s_wait_tensorcnt)
#define CDNA5_HAS_TDM 1
#endif
#endif
#ifndef CDNA5_HAS_TDM
#define CDNA5_HAS_TDM 0
#endif

static __device__ __forceinline__ unsigned short f2bf(float f) {
  unsigned int u = __float_as_uint(f);
  u += 0x7FFFu + ((u >> 16) & 1u);           // round-to-nearest-even
  return (unsigned short)(u >> 16);
}
static __device__ __forceinline__ unsigned int pack2bf(float a, float b) {
  return (unsigned int)f2bf(a) | ((unsigned int)f2bf(b) << 16);
}
// order-preserving float -> uint map for atomicMax-based segment max
static __device__ __forceinline__ unsigned int fenc(float x) {
  unsigned int u = __float_as_uint(x);
  return (u & 0x80000000u) ? ~u : (u | 0x80000000u);
}
static __device__ __forceinline__ float fdec(unsigned int u) {
  u = (u & 0x80000000u) ? (u & 0x7FFFFFFFu) : ~u;
  return __uint_as_float(u);
}

union BFrag { unsigned int u[8]; v16bf v; };

// ---------------------------------------------------------------------------
// Weight fusion: Wf[h*16+j, i] = sum_d rel[h,d,j] * W[h*16+d, i]  (bf16 out)
// rel == nullptr -> identity copy. bias fused likewise (f32).
// ---------------------------------------------------------------------------
__global__ __launch_bounds__(128) void fuse_kernel(
    const float* __restrict__ W, const float* __restrict__ b,
    const float* __restrict__ rel,
    unsigned short* __restrict__ Wf, float* __restrict__ bf) {
  int j = blockIdx.x;        // fused output row 0..127
  int i = threadIdx.x;       // input column 0..127
  int h = j >> 4, jj = j & 15;
  float acc;
  if (rel) {
    acc = 0.f;
    #pragma unroll
    for (int d = 0; d < 16; ++d)
      acc += rel[h * 256 + d * 16 + jj] * W[(h * 16 + d) * 128 + i];
  } else {
    acc = W[j * 128 + i];
  }
  Wf[j * 128 + i] = f2bf(acc);
  if (i == 0) {
    float bb;
    if (rel) {
      bb = 0.f;
      #pragma unroll
      for (int d = 0; d < 16; ++d) bb += rel[h * 256 + d * 16 + jj] * b[h * 16 + d];
    } else {
      bb = b[j];
    }
    bf[j] = bb;
  }
}

// ---------------------------------------------------------------------------
// GEMM: Y[M,128] = X[M,128] @ Wf^T + bias   (Wf row-major [n][k], bf16)
// 256 threads = 8 waves; each wave computes a 16x128 output strip with
// v_wmma_f32_16x16x32_bf16 (4 K-steps). Requires M % 16 == 0 (holds here).
// Weight tile (32KB) staged into LDS by the Tensor Data Mover; an asm pointer
// escape makes the TDM's LDS write visible to the compiler's alias analysis.
// ---------------------------------------------------------------------------
__global__ __launch_bounds__(256) void gemm_bf16_kernel(
    const float* __restrict__ X, const unsigned short* __restrict__ Wbf,
    const float* __restrict__ bias, float* __restrict__ Y, int M) {
  __shared__ unsigned short wlds[128 * 128];  // 32 KB, LDS offset 0
  const int t = threadIdx.x;

#if CDNA5_HAS_TDM
  if (t < 32) {
    // D# for a contiguous 32KB copy: 2D tensor, data_size=8B,
    // tensor_dim0 = tile_dim0 = 4096, tensor_dim1 = tile_dim1 = 1.
    unsigned long long ga = (unsigned long long)Wbf;
    u32x4 g0 = { 1u,                                  // count=1 (valid), user mode
                 0u,                                  // lds_addr = 0
                 (unsigned int)ga,                    // global_addr[31:0]
                 (((unsigned int)(ga >> 32)) & 0x01FFFFFFu) | 0x80000000u }; // addr[56:32] | type=2
    i32x8 g1 = { 0x00030000,                          // data_size=3 (8B), wg_mask=0
                 (int)(4096u << 16),                  // tensor_dim0 low16 -> bits[63:48]
                 1 << 16,                             // tensor_dim1 low16 -> bits[95:80]
                 (int)(4096u << 16),                  // tile_dim0 -> bits[127:112]
                 1,                                   // tile_dim1 -> bits[143:128]
                 4096,                                // tensor_dim0_stride low32
                 0, 0 };
    i32x4 z4 = { 0, 0, 0, 0 };
#if __clang_major__ >= 23
    i32x8 z8 = { 0, 0, 0, 0, 0, 0, 0, 0 };
    __builtin_amdgcn_tensor_load_to_lds(g0, g1, z4, z4, z8, 0);
#else
    __builtin_amdgcn_tensor_load_to_lds(g0, g1, z4, z4, 0);
#endif
    __builtin_amdgcn_s_wait_tensorcnt(0);
  }
  __syncthreads();
  // The TDM write to LDS is invisible to the compiler; escape the pointer
  // through an asm with a memory clobber so reads of wlds are not folded
  // to undef (round-2 regression: B fragments were DCE'd otherwise).
  {
    unsigned short* wp = &wlds[0];
    asm volatile("" : "+v"(wp) : : "memory");
  }
#else
  {
    const uint4* src = (const uint4*)Wbf;     // 2048 x 16B
    uint4* dst = (uint4*)wlds;
    for (int i = t; i < 2048; i += 256) dst[i] = src[i];
  }
  __syncthreads();
#endif

  const int wave = t >> 5;
  const int lane = t & 31;
  const int tile = blockIdx.x * 8 + wave;     // 16-row tile index
  if (tile >= (M >> 4)) return;

  const int m  = lane & 15;
  const int kg = lane >> 4;
  const float* row = X + (size_t)(tile * 16 + m) * 128;

  // warm L2 for the next block's A-tile rows (global_prefetch_b8)
  if (tile + 8 < (M >> 4)) __builtin_prefetch(row + (size_t)8 * 16 * 128, 0, 1);

  // A fragments (16x32 bf16 per K-tile), per ISA 16-bit A layout
  BFrag afrag[4];
  #pragma unroll
  for (int kt = 0; kt < 4; ++kt) {
    #pragma unroll
    for (int p = 0; p < 8; ++p) {
      int k0 = kt * 32 + 2 * p + ((p >= 4) ? 8 : 0) + 8 * kg;
      float2 f = *(const float2*)(row + k0);
      afrag[kt].u[p] = pack2bf(f.x, f.y);
    }
  }

  float* out = Y + (size_t)(tile * 16) * 128;
  for (int nt = 0; nt < 8; ++nt) {
    v8f c = {};
    #pragma unroll
    for (int kt = 0; kt < 4; ++kt) {
      BFrag bfrag;
      const int n = nt * 16 + m;
      #pragma unroll
      for (int p = 0; p < 8; ++p) {
        int k0 = kt * 32 + 2 * p + ((p >= 4) ? 8 : 0) + 8 * kg;   // B: same K order as A (table order)
        bfrag.u[p] = *(const unsigned int*)(&wlds[n * 128 + k0]);
      }
      c = __builtin_amdgcn_wmma_f32_16x16x32_bf16(
          false, afrag[kt].v, false, bfrag.v, (short)0, c, false, false);
    }
    const int ncol = nt * 16 + m;
    const float bv = bias[ncol];
    #pragma unroll
    for (int j = 0; j < 8; ++j) {
      int mrow = j + 8 * kg;                  // C/D: VGPR j -> M=j (+8 for lanes>=16)
      out[(size_t)mrow * 128 + ncol] = c[j] + bv;
    }
  }
}

// ---------------------------------------------------------------------------
// Small utility kernels
// ---------------------------------------------------------------------------
__global__ void fill_f32(float* p, float v, long long n) {
  long long i = (long long)blockIdx.x * blockDim.x + threadIdx.x;
  long long st = (long long)gridDim.x * blockDim.x;
  for (; i < n; i += st) p[i] = v;
}
__global__ void fill_enc(unsigned int* p, float v, long long n) {
  long long i = (long long)blockIdx.x * blockDim.x + threadIdx.x;
  long long st = (long long)gridDim.x * blockDim.x;
  unsigned int e = fenc(v);
  for (; i < n; i += st) p[i] = e;
}
__global__ void scatter_deg(const int* __restrict__ idx, float* __restrict__ deg, int E) {
  int g = blockIdx.x * blockDim.x + threadIdx.x;
  if (g < E) atomicAdd(&deg[idx[g]], 1.f);
}
__global__ void scatter_mask(const int* __restrict__ idx, int* __restrict__ mask, int E) {
  int g = blockIdx.x * blockDim.x + threadIdx.x;
  if (g < E) mask[idx[g]] = 1;
}
__global__ void count_zero(const float* __restrict__ deg, unsigned int* cnt, int N) {
  int g = blockIdx.x * blockDim.x + threadIdx.x;
  if (g < N && deg[g] == 0.f) atomicAdd(cnt, 1u);
}
__global__ void norms_kernel(const float* __restrict__ X, float* __restrict__ inv, int N) {
  int wid = (blockIdx.x * blockDim.x + threadIdx.x) >> 5;
  int lane = threadIdx.x & 31;
  if (wid >= N) return;
  float4 v = ((const float4*)(X + (size_t)wid * 128))[lane];
  float s = v.x * v.x + v.y * v.y + v.z * v.z + v.w * v.w;
  for (int off = 16; off; off >>= 1) s += __shfl_xor(s, off, 32);
  if (lane == 0) inv[wid] = 1.0f / fmaxf(sqrtf(s), 1e-12f);
}
// blend relation-2 K/V with conversation context on masked nodes (post-transform,
// valid by linearity of the relation transform)
__global__ void blend_kernel(float* __restrict__ K2t, float* __restrict__ V2t,
                             const float* __restrict__ CKt, const float* __restrict__ CVt,
                             const int* __restrict__ mask, const float* __restrict__ conv_w,
                             long long n) {
  long long g = (long long)blockIdx.x * blockDim.x + threadIdx.x;
  if (g >= n) return;
  int node = (int)(g >> 7);
  if (!mask[node]) return;
  float w = 1.f / (1.f + __expf(-conv_w[0]));
  K2t[g] = (1.f - w) * K2t[g] + w * CKt[g];
  V2t[g] = (1.f - w) * V2t[g] + w * CVt[g];
}

// ---------------------------------------------------------------------------
// Edge attention, softmax, aggregation. 8 lanes per edge (one head each).
// ---------------------------------------------------------------------------
__global__ __launch_bounds__(256) void attn1_kernel(
    const float* __restrict__ Qt, const float* __restrict__ Kt,
    const float* __restrict__ hu, const float* __restrict__ hc,
    const float* __restrict__ inu, const float* __restrict__ inc_,
    const int* __restrict__ src, const int* __restrict__ dst,
    const float* __restrict__ sim_boost, const unsigned int* __restrict__ zc,
    float* __restrict__ attn, unsigned int* __restrict__ mx, int E) {
  int gid = blockIdx.x * 256 + threadIdx.x;
  int e = gid >> 3, h = gid & 7;
  if (e >= E) return;
  int s = src[e], d = dst[e];
  const float4* q = (const float4*)(Qt + (size_t)d * 128 + h * 16);
  const float4* k = (const float4*)(Kt + (size_t)s * 128 + h * 16);
  float acc = 0.f;
  #pragma unroll
  for (int i = 0; i < 4; ++i) {
    float4 a = q[i], b = k[i];
    acc += a.x * b.x + a.y * b.y + a.z * b.z + a.w * b.w;
  }
  // cosine similarity over full 128 dims, split across the 8 head-lanes
  const float4* us = (const float4*)(hu + (size_t)s * 128 + h * 16);
  const float4* cs = (const float4*)(hc + (size_t)d * 128 + h * 16);
  float cacc = 0.f;
  #pragma unroll
  for (int i = 0; i < 4; ++i) {
    float4 a = us[i], b = cs[i];
    cacc += a.x * b.x + a.y * b.y + a.z * b.z + a.w * b.w;
  }
  cacc += __shfl_xor(cacc, 1, 32);
  cacc += __shfl_xor(cacc, 2, 32);
  cacc += __shfl_xor(cacc, 4, 32);
  float sparsity = (float)(*zc) / (float)NC;
  float boost = (sparsity > 0.3f) ? (sim_boost[0] * cacc * inu[s] * inc_[d]) : 0.f;
  float a = acc * 0.25f + boost;              // 1/sqrt(DK) = 0.25
  attn[(size_t)e * 8 + h] = a;
  atomicMax(mx + (size_t)d * 8 + h, fenc(a));
}

__global__ __launch_bounds__(256) void attn2_kernel(
    const float* __restrict__ Qt, const float* __restrict__ Kt,
    const float* __restrict__ EF,
    const int* __restrict__ src, const int* __restrict__ dst,
    float* __restrict__ attn, unsigned int* __restrict__ mx, int E) {
  int gid = blockIdx.x * 256 + threadIdx.x;
  int e = gid >> 3, h = gid & 7;
  if (e >= E) return;
  int s = src[e], d = dst[e];
  const float4* q = (const float4*)(Qt + (size_t)d * 128 + h * 16);
  const float4* k = (const float4*)(Kt + (size_t)s * 128 + h * 16);
  const float4* f = (const float4*)(EF + (size_t)e * 128 + h * 16);
  float acc = 0.f;
  #pragma unroll
  for (int i = 0; i < 4; ++i) {
    float4 qa = q[i], ka = k[i], fa = f[i];
    acc += qa.x * (ka.x + fa.x) + qa.y * (ka.y + fa.y) +
           qa.z * (ka.z + fa.z) + qa.w * (ka.w + fa.w);
  }
  float a = acc * 0.25f;
  attn[(size_t)e * 8 + h] = a;
  atomicMax(mx + (size_t)d * 8 + h, fenc(a));
}

__global__ __launch_bounds__(256) void exp_sum_kernel(
    float* __restrict__ attn, const unsigned int* __restrict__ mx,
    float* __restrict__ sum, const int* __restrict__ dst, int E) {
  int gid = blockIdx.x * 256 + threadIdx.x;
  int e = gid >> 3, h = gid & 7;
  if (e >= E) return;
  int d = dst[e];
  float m = fdec(mx[(size_t)d * 8 + h]);
  float ev = __expf(attn[(size_t)e * 8 + h] - m);
  attn[(size_t)e * 8 + h] = ev;
  atomicAdd(sum + (size_t)d * 8 + h, ev);
}

__global__ __launch_bounds__(256) void agg_kernel(
    const float* __restrict__ attn, const float* __restrict__ sum,
    const float* __restrict__ Vt, const int* __restrict__ src,
    const int* __restrict__ dst, float* __restrict__ agg, int E) {
  int gid = blockIdx.x * 256 + threadIdx.x;
  int e = gid >> 3, h = gid & 7;
  if (e >= E) return;
  int s = src[e], d = dst[e];
  float a = attn[(size_t)e * 8 + h] / sum[(size_t)d * 8 + h];
  const float* v = Vt + (size_t)s * 128 + h * 16;
  float* o = agg + (size_t)d * 128 + h * 16;
  #pragma unroll
  for (int i = 0; i < 16; ++i) atomicAdd(o + i, a * v[i]);
}

// LayerNorm (wave per row) + context accumulation over nonzero-degree rows
__global__ __launch_bounds__(256) void ln_kernel(
    const float* __restrict__ X, const float* __restrict__ g,
    const float* __restrict__ b, const float* __restrict__ deg,
    float* __restrict__ out, float* __restrict__ ctxsum,
    unsigned int* __restrict__ cnt, int N) {
  int wid = (blockIdx.x * 256 + threadIdx.x) >> 5;
  int lane = threadIdx.x & 31;
  if (wid >= N) return;
  float4 x = ((const float4*)(X + (size_t)wid * 128))[lane];
  float s = x.x + x.y + x.z + x.w;
  float ss = x.x * x.x + x.y * x.y + x.z * x.z + x.w * x.w;
  for (int off = 16; off; off >>= 1) {
    s += __shfl_xor(s, off, 32);
    ss += __shfl_xor(ss, off, 32);
  }
  float mu = s * (1.f / 128.f);
  float var = ss * (1.f / 128.f) - mu * mu;
  float rstd = rsqrtf(var + 1e-5f);
  float4 gg = ((const float4*)g)[lane];
  float4 bb = ((const float4*)b)[lane];
  float4 y;
  y.x = (x.x - mu) * rstd * gg.x + bb.x;
  y.y = (x.y - mu) * rstd * gg.y + bb.y;
  y.z = (x.z - mu) * rstd * gg.z + bb.z;
  y.w = (x.w - mu) * rstd * gg.w + bb.w;
  ((float4*)(out + (size_t)wid * 128))[lane] = y;
  if (deg[wid] != 0.f) {
    int i0 = lane * 4;
    atomicAdd(ctxsum + i0 + 0, y.x);
    atomicAdd(ctxsum + i0 + 1, y.y);
    atomicAdd(ctxsum + i0 + 2, y.z);
    atomicAdd(ctxsum + i0 + 3, y.w);
    if (lane == 0) atomicAdd(cnt, 1u);
  }
}

// zero-degree fixup: rows with deg==0 get 0.9*init + 0.1*ctx (ctx = mean over
// nonzero rows of the LN result); nonzero rows already hold the final value.
__global__ void fix_kernel(const float* __restrict__ init, const float* __restrict__ deg,
                           const float* __restrict__ ctxsum, const unsigned int* __restrict__ cnt,
                           float* __restrict__ out, long long n) {
  long long g = (long long)blockIdx.x * blockDim.x + threadIdx.x;
  if (g >= n) return;
  int node = (int)(g >> 7), i = (int)(g & 127);
  if (deg[node] != 0.f) return;
  float c = (float)(*cnt);
  float ini = init[g];
  out[g] = (c > 0.f) ? (0.9f * ini + 0.1f * (ctxsum[i] / c)) : ini;
}

static inline int cdiv(long long a, long long b) { return (int)((a + b - 1) / b); }

extern "C" void kernel_launch(void* const* d_in, const int* in_sizes, int n_in,
                              void* d_out, int out_size, void* d_ws, size_t ws_size,
                              hipStream_t stream) {
  (void)in_sizes; (void)n_in; (void)out_size; (void)ws_size;
  const float* h_user    = (const float*)d_in[0];
  const float* h_comment = (const float*)d_in[1];
  const float* edge_h    = (const float*)d_in[2];
  const float* user_ctx  = (const float*)d_in[3];
  const int*   src1      = (const int*)d_in[4];
  const int*   dst1      = (const int*)d_in[5];
  const int*   src2      = (const int*)d_in[6];
  const int*   dst2      = (const int*)d_in[7];
  // params flattened in insertion order starting at index 8
  const float* k_user_W    = (const float*)d_in[8];
  const float* k_user_b    = (const float*)d_in[9];
  const float* q_user_W    = (const float*)d_in[10];
  const float* q_user_b    = (const float*)d_in[11];
  const float* v_user_W    = (const float*)d_in[12];
  const float* v_user_b    = (const float*)d_in[13];
  const float* a_user_W    = (const float*)d_in[14];
  const float* a_user_b    = (const float*)d_in[15];
  const float* ln_user_g   = (const float*)d_in[16];
  const float* ln_user_b   = (const float*)d_in[17];
  const float* q_comment_W = (const float*)d_in[20];
  const float* q_comment_b = (const float*)d_in[21];
  const float* a_comment_W = (const float*)d_in[24];
  const float* a_comment_b = (const float*)d_in[25];
  const float* ln_comment_g= (const float*)d_in[26];
  const float* ln_comment_b= (const float*)d_in[27];
  const float* rel_posts_k = (const float*)d_in[28];
  const float* rel_posts_q = (const float*)d_in[29];
  const float* rel_posts_v = (const float*)d_in[30];
  const float* rel_ucu_k   = (const float*)d_in[31];
  const float* rel_ucu_q   = (const float*)d_in[32];
  const float* rel_ucu_v   = (const float*)d_in[33];
  const float* edge_W      = (const float*)d_in[34];
  const float* edge_b      = (const float*)d_in[35];
  const float* convk_W     = (const float*)d_in[36];
  const float* convk_b     = (const float*)d_in[37];
  const float* convv_W     = (const float*)d_in[38];
  const float* convv_b     = (const float*)d_in[39];
  const float* conv_w      = (const float*)d_in[40];
  const float* sim_boost   = (const float*)d_in[41];

  char* base = (char*)d_ws;
  size_t off = 0;
  auto alloc = [&](size_t bytes) -> char* {
    char* p = base + off;
    off += (bytes + 255) & ~(size_t)255;
    return p;
  };

  unsigned short* fusedW = (unsigned short*)alloc(11 * 128 * 128 * sizeof(unsigned short));
  float* fusedB = (float*)alloc(11 * 128 * sizeof(float));
  float* K1t = (float*)alloc((size_t)NU * 128 * 4);
  float* Q1t = (float*)alloc((size_t)NC * 128 * 4);
  float* V1t = (float*)alloc((size_t)NU * 128 * 4);
  float* K2t = (float*)alloc((size_t)NU * 128 * 4);
  float* Q2t = (float*)alloc((size_t)NU * 128 * 4);
  float* V2t = (float*)alloc((size_t)NU * 128 * 4);
  float* CKt = (float*)alloc((size_t)NU * 128 * 4);
  float* CVt = (float*)alloc((size_t)NU * 128 * 4);
  float* EF  = (float*)alloc((size_t)E2N * 128 * 4);
  float* at1 = (float*)alloc((size_t)E1N * 8 * 4);
  float* at2 = (float*)alloc((size_t)E2N * 8 * 4);
  float* inu = (float*)alloc((size_t)NU * 4);
  float* inc_ = (float*)alloc((size_t)NC * 4);
  // ---- zero-initialized group (contiguous) ----
  size_t z0 = off;
  float* sum1  = (float*)alloc((size_t)NC * 8 * 4);
  float* sum2  = (float*)alloc((size_t)NU * 8 * 4);
  float* agg_c = (float*)alloc((size_t)NC * 128 * 4);
  float* agg_u = (float*)alloc((size_t)NU * 128 * 4);
  float* deg_u = (float*)alloc((size_t)NU * 4);
  float* deg_c = (float*)alloc((size_t)NC * 4);
  int*   mask  = (int*)alloc((size_t)NU * 4);
  float* ctx_u = (float*)alloc(128 * 4);
  float* ctx_c = (float*)alloc(128 * 4);
  unsigned int* cnts = (unsigned int*)alloc(256);  // [0]=cnt_u [1]=cnt_c [2]=zero_cnt
  size_t z1 = off;
  // ---- segment-max buffers (init to encoded -inf) ----
  size_t m0 = off;
  unsigned int* mx1 = (unsigned int*)alloc((size_t)NC * 8 * 4);
  unsigned int* mx2 = (unsigned int*)alloc((size_t)NU * 8 * 4);
  size_t m1 = off;
  // pre-LN output buffers alias EF (EF is dead by then; E2N*128 >= (NU+NC)*128)
  float* pre_c = EF;
  float* pre_u = EF + (size_t)NC * 128;

  float* out_u = (float*)d_out;
  float* out_c = out_u + (size_t)NU * 128;

  // 1) init scratch
  fill_f32<<<4096, 256, 0, stream>>>((float*)(base + z0), 0.f, (long long)((z1 - z0) / 4));
  fill_enc<<<1024, 256, 0, stream>>>((unsigned int*)(base + m0), -3.0e38f, (long long)((m1 - m0) / 4));

  // 2) fuse weights (projection ∘ relation transform), bf16
  auto fW = [&](int i) { return fusedW + (size_t)i * 128 * 128; };
  auto fB = [&](int i) { return fusedB + (size_t)i * 128; };
  fuse_kernel<<<128, 128, 0, stream>>>(k_user_W,    k_user_b,    rel_posts_k, fW(0),  fB(0));
  fuse_kernel<<<128, 128, 0, stream>>>(q_comment_W, q_comment_b, rel_posts_q, fW(1),  fB(1));
  fuse_kernel<<<128, 128, 0, stream>>>(v_user_W,    v_user_b,    rel_posts_v, fW(2),  fB(2));
  fuse_kernel<<<128, 128, 0, stream>>>(k_user_W,    k_user_b,    rel_ucu_k,   fW(3),  fB(3));
  fuse_kernel<<<128, 128, 0, stream>>>(q_user_W,    q_user_b,    rel_ucu_q,   fW(4),  fB(4));
  fuse_kernel<<<128, 128, 0, stream>>>(v_user_W,    v_user_b,    rel_ucu_v,   fW(5),  fB(5));
  fuse_kernel<<<128, 128, 0, stream>>>(convk_W,     convk_b,     rel_ucu_k,   fW(6),  fB(6));
  fuse_kernel<<<128, 128, 0, stream>>>(convv_W,     convv_b,     rel_ucu_v,   fW(7),  fB(7));
  fuse_kernel<<<128, 128, 0, stream>>>(edge_W,      edge_b,      nullptr,     fW(8),  fB(8));
  fuse_kernel<<<128, 128, 0, stream>>>(a_user_W,    a_user_b,    nullptr,     fW(9),  fB(9));
  fuse_kernel<<<128, 128, 0, stream>>>(a_comment_W, a_comment_b, nullptr,     fW(10), fB(10));

  // 3) node/edge projections via bf16 WMMA GEMMs (weights staged by TDM)
  auto gemm = [&](const float* X, int w, float* Y, int M) {
    gemm_bf16_kernel<<<cdiv(M / 16, 8), 256, 0, stream>>>(X, fW(w), fB(w), Y, M);
  };
  gemm(h_user,    0, K1t, NU);
  gemm(h_comment, 1, Q1t, NC);
  gemm(h_user,    2, V1t, NU);
  gemm(h_user,    3, K2t, NU);
  gemm(h_user,    4, Q2t, NU);
  gemm(h_user,    5, V2t, NU);
  gemm(user_ctx,  6, CKt, NU);
  gemm(user_ctx,  7, CVt, NU);
  gemm(edge_h,    8, EF,  E2N);

  // 4) degrees, mask, norms, sparsity count
  scatter_deg<<<cdiv(E1N, 256), 256, 0, stream>>>(src1, deg_u, E1N);
  scatter_deg<<<cdiv(E2N, 256), 256, 0, stream>>>(src2, deg_u, E2N);
  scatter_deg<<<cdiv(E2N, 256), 256, 0, stream>>>(dst2, deg_u, E2N);
  scatter_deg<<<cdiv(E1N, 256), 256, 0, stream>>>(dst1, deg_c, E1N);
  scatter_mask<<<cdiv(E2N, 256), 256, 0, stream>>>(src2, mask, E2N);
  count_zero<<<cdiv(NC, 256), 256, 0, stream>>>(deg_c, cnts + 2, NC);
  norms_kernel<<<cdiv((long long)NU * 32, 256), 256, 0, stream>>>(h_user, inu, NU);
  norms_kernel<<<cdiv((long long)NC * 32, 256), 256, 0, stream>>>(h_comment, inc_, NC);

  // 5) conversation blend on masked users (post relation transform)
  blend_kernel<<<cdiv((long long)NU * 128, 256), 256, 0, stream>>>(
      K2t, V2t, CKt, CVt, mask, conv_w, (long long)NU * 128);

  // 6) relation 1: attention -> softmax -> aggregation into agg_c
  attn1_kernel<<<cdiv((long long)E1N * 8, 256), 256, 0, stream>>>(
      Q1t, K1t, h_user, h_comment, inu, inc_, src1, dst1, sim_boost, cnts + 2, at1, mx1, E1N);
  exp_sum_kernel<<<cdiv((long long)E1N * 8, 256), 256, 0, stream>>>(at1, mx1, sum1, dst1, E1N);
  agg_kernel<<<cdiv((long long)E1N * 8, 256), 256, 0, stream>>>(at1, sum1, V1t, src1, dst1, agg_c, E1N);

  // 7) relation 2: attention (with edge features) -> softmax -> agg into agg_u
  attn2_kernel<<<cdiv((long long)E2N * 8, 256), 256, 0, stream>>>(
      Q2t, K2t, EF, src2, dst2, at2, mx2, E2N);
  exp_sum_kernel<<<cdiv((long long)E2N * 8, 256), 256, 0, stream>>>(at2, mx2, sum2, dst2, E2N);
  agg_kernel<<<cdiv((long long)E2N * 8, 256), 256, 0, stream>>>(at2, sum2, V2t, src2, dst2, agg_u, E2N);

  // 8) output linears (WMMA) -> LayerNorm (+ ctx accumulation) -> zero-deg fixup
  gemm(agg_u, 9,  pre_u, NU);
  gemm(agg_c, 10, pre_c, NC);
  ln_kernel<<<cdiv(NU, 8), 256, 0, stream>>>(pre_u, ln_user_g, ln_user_b, deg_u,
                                             out_u, ctx_u, cnts + 0, NU);
  ln_kernel<<<cdiv(NC, 8), 256, 0, stream>>>(pre_c, ln_comment_g, ln_comment_b, deg_c,
                                             out_c, ctx_c, cnts + 1, NC);
  fix_kernel<<<cdiv((long long)NU * 128, 256), 256, 0, stream>>>(
      h_user, deg_u, ctx_u, cnts + 0, out_u, (long long)NU * 128);
  fix_kernel<<<cdiv((long long)NC * 128, 256), 256, 0, stream>>>(
      h_comment, deg_c, ctx_c, cnts + 1, out_c, (long long)NC * 128);
}
